// MGDiffMAttentionReductionLayer_10608569221309
// MI455X (gfx1250) — compile-verified, matrix operand order checked by reading, not run on hardware
//
#include <hip/hip_runtime.h>
#include <hip/hip_bf16.h>

// ---------------------------------------------------------------------------
// MI455X (gfx1250) fully fused per-token kernel, v3.
// - All GEMMs on v_wmma_f32_16x16x32_bf16 (wave32 WMMA, f32 accumulate)
// - 32-token tiles; B tiles register-resident across M tiles (6x reuse in
//   qkv/o-proj, 2x elsewhere) to cut L2 weight traffic
// - LDS buffers overlapped by lifetime: 160 KB total -> 2 workgroups/WGP
// - __launch_bounds__(256,4): cap VGPRs to <=256/wave so 4 waves/SIMD fit,
//   matching the 2-WG LDS budget (v2 emitted s_set_vgpr_msb => >256 VGPRs)
// ---------------------------------------------------------------------------

typedef __attribute__((ext_vector_type(16))) __bf16 v16bf;
typedef __attribute__((ext_vector_type(8)))  __bf16 v8bf;
typedef __attribute__((ext_vector_type(8)))  float  v8f;

#define TOKS   32          // tokens per workgroup (two WMMA M tiles)
#define CC     128
#define GG     3
#define ROWS   (GG * TOKS) // 96 token x group rows
#define KIN    448         // 64 + 128 + 256

// swizzled-weight element offsets inside workspace (bf16 elements)
#define OFF_WP0 0
#define OFF_WP2 8192
#define OFF_WSK 40960
#define OFF_WQ  98304
#define OFF_WK  114688
#define OFF_WV  131072
#define OFF_WO  147456
#define OFF_W1  163840
#define OFF_W2  212992
#define WSW_ELEMS 229376
#define WS_MOD_BYTE_OFF (WSW_ELEMS * 2)   // 458752, 16B aligned

// ---------------------------------------------------------------------------
// Kernel 1: AdaLN modulation tables.  mod[0..767] = ln1 (g-major rows of 256:
// scale[0:128], shift[128:256]); mod[768..1535] = ln2.
// ---------------------------------------------------------------------------
__global__ void k_mod(const float* __restrict__ emb, const int* __restrict__ gidx,
                      const float* __restrict__ W1, const float* __restrict__ B1,
                      const float* __restrict__ W2, const float* __restrict__ B2,
                      float* __restrict__ mod) {
    int j = blockIdx.x * blockDim.x + threadIdx.x;
    if (j >= 2 * GG * 256) return;
    int which = j / (GG * 256);
    int rem   = j % (GG * 256);
    int g = rem / 256, col = rem % 256;
    const float* W  = which ? W2 : W1;
    const float* Bv = which ? B2 : B1;
    const float* e  = emb + gidx[g] * 64;
    float s = 0.f;
#pragma unroll 8
    for (int i = 0; i < 64; ++i) s += e[i] * W[i * 256 + col];
    mod[j] = s + Bv[col];
}

// ---------------------------------------------------------------------------
// Kernel 2: f32 weights -> bf16, pre-swizzled into WMMA-B tile layout.
// Tile (kt,nt): lane L element e holds W[kt*32 + kmap][nt*16 + (L&15)],
// kmap = e + 8*(e>=8) + 8*(L>=16).  512 bf16 per tile, 16 contiguous per lane.
// ---------------------------------------------------------------------------
__global__ void k_swz(const float* __restrict__ wp0, const float* __restrict__ wp2,
                      const float* __restrict__ wsk, const float* __restrict__ wq,
                      const float* __restrict__ wk,  const float* __restrict__ wv,
                      const float* __restrict__ wo,  const float* __restrict__ w1,
                      const float* __restrict__ w2,  __bf16* __restrict__ dst) {
    const int tiles[9] = {16, 64, 112, 32, 32, 32, 32, 96, 32};
    const int offs [9] = {OFF_WP0, OFF_WP2, OFF_WSK, OFF_WQ, OFF_WK,
                          OFF_WV,  OFF_WO,  OFF_W1,  OFF_W2};
    int b = blockIdx.x, m = 0, base = 0;
    while (b >= base + tiles[m]) { base += tiles[m]; ++m; }
    int lt = b - base;
    const float* src = (m == 0) ? wp0 : (m == 1) ? wp2 : (m == 2) ? wsk :
                       (m == 3) ? wq  : (m == 4) ? wk  : (m == 5) ? wv :
                       (m == 6) ? wo  : (m == 7) ? w1  : w2;
    int kt = lt >> 3, nt = lt & 7;
    int tid = threadIdx.x;              // 512 threads: L = tid>>4, e = tid&15
    int L = tid >> 4, e = tid & 15;
    int kmap = e + ((e >= 8) ? 8 : 0) + ((L >= 16) ? 8 : 0);
    int k = kt * 32 + kmap;
    int n = nt * 16 + (L & 15);
    dst[offs[m] + lt * 512 + L * 16 + e] = (__bf16)src[k * 128 + n];
}

// ---------------------------------------------------------------------------
// WMMA operand helpers
// ---------------------------------------------------------------------------
__device__ __forceinline__ v16bf ldsA(const __bf16* buf, int ld, int r0, int kc) {
    int lane = threadIdx.x & 31;
    int M = lane & 15, h = lane >> 4;
    const __bf16* p = buf + (r0 + M) * ld + kc + 8 * h;
    v8bf lo = *(const v8bf*)p;         // K = 8h .. 8h+7
    v8bf hi = *(const v8bf*)(p + 16);  // K = 16+8h ..
    v16bf a;
#pragma unroll
    for (int i = 0; i < 8; ++i) { a[i] = lo[i]; a[8 + i] = hi[i]; }
    return a;
}

__device__ __forceinline__ v16bf ldB(const __bf16* w, int tile) {
    int lane = threadIdx.x & 31;
    return *(const v16bf*)(w + (size_t)tile * 512 + lane * 16);
}

__device__ __forceinline__ v8f wmma_acc(v16bf a, v16bf b, v8f c) {
    return __builtin_amdgcn_wmma_f32_16x16x32_bf16(false, a, false, b,
                                                   (short)0, c, false, false);
}

// D layout: VGPR r -> M = r + 8*(lane>>4), N = lane&15.
__device__ __forceinline__ void stD(float* buf, int ld, int r0, int c0, v8f acc) {
    int lane = threadIdx.x & 31;
    int n = lane & 15, h = lane >> 4;
#pragma unroll
    for (int r = 0; r < 8; ++r) buf[(r0 + r + 8 * h) * ld + c0 + n] = acc[r];
}
__device__ __forceinline__ void stD_bias(float* buf, int ld, int r0, int c0, v8f acc,
                                         const float* __restrict__ bias) {
    int lane = threadIdx.x & 31;
    int n = lane & 15, h = lane >> 4;
#pragma unroll
    for (int r = 0; r < 8; ++r)
        buf[(r0 + r + 8 * h) * ld + c0 + n] = acc[r] + bias[c0 + n];
}
__device__ __forceinline__ void stDbf(__bf16* buf, int ld, int r0, int c0, v8f acc) {
    int lane = threadIdx.x & 31;
    int n = lane & 15, h = lane >> 4;
#pragma unroll
    for (int r = 0; r < 8; ++r)
        buf[(r0 + r + 8 * h) * ld + c0 + n] = (__bf16)acc[r];
}

// ---------------------------------------------------------------------------
// Kernel 3: fused per-token-tile network (32 tokens / workgroup, 8 waves).
// ---------------------------------------------------------------------------
__global__ void __launch_bounds__(256, 4)
k_fused(const float* __restrict__ x0, const float* __restrict__ x1,
        const float* __restrict__ x2,
        const unsigned char* __restrict__ mk0, const unsigned char* __restrict__ mk1,
        const unsigned char* __restrict__ mk2,
        const __bf16* __restrict__ wsw, const float* __restrict__ mod,
        const float* __restrict__ bo,  const float* __restrict__ b_skip,
        const float* __restrict__ gamma, const float* __restrict__ gamma_mlp,
        const float* __restrict__ b1,  const float* __restrict__ b2,
        float* __restrict__ out, int T) {
    // Union region (disjoint lifetimes, barrier-separated):
    //   phase 0/A : sInA  = bf16 [32][448] input concat  (28672 B)
    //   phase C/D : sQ/K/V = bf16 [96][128] each          (73728 B)
    //   phase F/G : sXn2  = bf16 [32][384]                (24576 B)
    __shared__ alignas(16) unsigned char sU[ROWS * CC * 2 * 3];   // 73728
    __shared__ float  sXg  [ROWS * CC];   // 49152: stacked proj, then residual x2
    __shared__ float  sSkip[TOKS * CC];   // 16384: x_skip_outer
    __shared__ __bf16 sXn1 [ROWS * CC];   // 24576: xn1 -> attn out -> MLP hidden
    __bf16* sInA = (__bf16*)sU;                      // ld KIN
    __bf16* sQb  = (__bf16*)sU;                      // ld CC
    __bf16* sKb  = (__bf16*)(sU + ROWS * CC * 2);
    __bf16* sVb  = (__bf16*)(sU + ROWS * CC * 4);
    __bf16* sXn2 = (__bf16*)sU;                      // ld GG*CC
    __bf16* sO   = sXn1;                             // attn output (after C)
    __bf16* sH   = sXn1;                             // MLP hidden (after F)

    const int tid  = threadIdx.x;
    const int wave = tid >> 5;
    const int lane = tid & 31;
    const int tok0 = blockIdx.x * TOKS;

    // ---- Stage 0: stage inputs into LDS (bf16 concat row [x0|x1|x2]); keep
    //      group-1 (identity) projection in f32 (sXg rows 32..63).
    for (int i = tid; i < TOKS * KIN; i += 256) {
        int t = i / KIN, c = i % KIN;
        int tok = tok0 + t;
        float v;
        if (c < 64)       v = x0[(size_t)tok * 64  + c];
        else if (c < 192) { v = x1[(size_t)tok * 128 + (c - 64)];
                            sXg[(TOKS + t) * CC + (c - 64)] = v; }
        else              v = x2[(size_t)tok * 256 + (c - 192)];
        sInA[t * KIN + c] = (__bf16)v;
    }
    __syncthreads();

    // ---- Stage A: skip GEMM (K=448) + proj0 (K=64) + proj2 (K=256).
    //      B loaded once per kt, reused over 2 token M-tiles.
    {
        int nt = wave;
        v8f s0 = {}, s1 = {};
        for (int kt = 0; kt < 14; ++kt) {            // unroll: compiler's choice
            v16bf b = ldB(wsw + OFF_WSK, kt * 8 + nt);
            s0 = wmma_acc(ldsA(sInA, KIN, 0,  kt * 32), b, s0);
            s1 = wmma_acc(ldsA(sInA, KIN, 16, kt * 32), b, s1);
        }
        stD_bias(sSkip, CC, 0,  nt * 16, s0, b_skip);
        stD_bias(sSkip, CC, 16, nt * 16, s1, b_skip);

        v8f p0 = {}, p1 = {};
#pragma unroll
        for (int kt = 0; kt < 2; ++kt) {
            v16bf b = ldB(wsw + OFF_WP0, kt * 8 + nt);
            p0 = wmma_acc(ldsA(sInA, KIN, 0,  kt * 32), b, p0);
            p1 = wmma_acc(ldsA(sInA, KIN, 16, kt * 32), b, p1);
        }
        stD(sXg, CC, 0,  nt * 16, p0);               // group 0, tokens 0..15
        stD(sXg, CC, 16, nt * 16, p1);               // group 0, tokens 16..31

        v8f q0 = {}, q1 = {};
#pragma unroll 4
        for (int kt = 0; kt < 8; ++kt) {
            v16bf b = ldB(wsw + OFF_WP2, kt * 8 + nt);
            q0 = wmma_acc(ldsA(sInA, KIN, 0,  192 + kt * 32), b, q0);
            q1 = wmma_acc(ldsA(sInA, KIN, 16, 192 + kt * 32), b, q1);
        }
        stD(sXg, CC, 2 * TOKS,      nt * 16, q0);    // group 2
        stD(sXg, CC, 2 * TOKS + 16, nt * 16, q1);
    }
    __syncthreads();

    // ---- Stage B: ada_ln1 over 384 values/token -> bf16 xn1 [g*32+t][c].
    {
        int t = tid >> 3, i = tid & 7;               // 8 threads per token
        float s = 0.f, s2 = 0.f;
#pragma unroll
        for (int g = 0; g < GG; ++g)
#pragma unroll
            for (int j = 0; j < 16; ++j) {
                float v = sXg[(g * TOKS + t) * CC + i * 16 + j];
                s += v; s2 += v * v;
            }
#pragma unroll
        for (int off = 4; off >= 1; off >>= 1) {
            s  += __shfl_xor(s,  off, 8);
            s2 += __shfl_xor(s2, off, 8);
        }
        float mu   = s * (1.f / 384.f);
        float var  = s2 * (1.f / 384.f) - mu * mu;
        float rstd = rsqrtf(var + 1e-5f);
#pragma unroll
        for (int g = 0; g < GG; ++g)
#pragma unroll
            for (int j = 0; j < 16; ++j) {
                int c = i * 16 + j;
                float v  = sXg[(g * TOKS + t) * CC + c];
                float sc = mod[g * 256 + c];
                float sh = mod[g * 256 + 128 + c];
                sXn1[(g * TOKS + t) * CC + c] =
                    (__bf16)(((v - mu) * rstd) * (1.f + sc) + sh);
            }
    }
    __syncthreads();

    // ---- Stage C: q/k/v = xn1(96x128) @ W(128x128).  Each wave owns one N
    //      tile; B tiles register-resident, reused over 6 M tiles.
    for (int mat = 0; mat < 3; ++mat) {
        const __bf16* W = wsw + OFF_WQ + mat * 16384;
        __bf16* dst = (mat == 0) ? sQb : (mat == 1) ? sKb : sVb;
        v8f acc[6] = {};
#pragma unroll
        for (int kt = 0; kt < 4; ++kt) {
            v16bf b = ldB(W, kt * 8 + wave);
#pragma unroll
            for (int mt = 0; mt < 6; ++mt)
                acc[mt] = wmma_acc(ldsA(sXn1, CC, mt * 16, kt * 32), b, acc[mt]);
        }
#pragma unroll
        for (int mt = 0; mt < 6; ++mt)
            stDbf(dst, CC, mt * 16, wave * 16, acc[mt]);
    }
    __syncthreads();

    // ---- Stage D: per-token 3x3 channel attention per 16-dim head.
    //      256 threads = 32 tokens x 8 heads.
    {
        int t = tid >> 3, hh = tid & 7;
        int tok = tok0 + t;
        float mneg[GG];
        mneg[0] = mk0[tok] ? 0.f : -1e9f;
        mneg[1] = mk1[tok] ? 0.f : -1e9f;
        mneg[2] = mk2[tok] ? 0.f : -1e9f;
#pragma unroll
        for (int qg = 0; qg < GG; ++qg) {
            float sc[GG];
#pragma unroll
            for (int kg = 0; kg < GG; ++kg) {
                float d = 0.f;
#pragma unroll
                for (int dd = 0; dd < 16; ++dd)
                    d += (float)sQb[(qg * TOKS + t) * CC + hh * 16 + dd] *
                         (float)sKb[(kg * TOKS + t) * CC + hh * 16 + dd];
                sc[kg] = d * 0.25f + mneg[kg];
            }
            float mx = fmaxf(sc[0], fmaxf(sc[1], sc[2]));
            float e0 = __expf(sc[0] - mx), e1 = __expf(sc[1] - mx),
                  e2 = __expf(sc[2] - mx);
            float rs = 1.f / (e0 + e1 + e2);
            e0 *= rs; e1 *= rs; e2 *= rs;
#pragma unroll
            for (int dd = 0; dd < 16; ++dd) {
                float o = e0 * (float)sVb[(0 * TOKS + t) * CC + hh * 16 + dd] +
                          e1 * (float)sVb[(1 * TOKS + t) * CC + hh * 16 + dd] +
                          e2 * (float)sVb[(2 * TOKS + t) * CC + hh * 16 + dd];
                sO[(qg * TOKS + t) * CC + hh * 16 + dd] = (__bf16)o;
            }
        }
    }
    __syncthreads();

    // ---- Stage E: o-proj (96x128 @ 128x128) + residual x2 = x_skip + g*att.
    {
        int nt = wave;
        v8f acc[6] = {};
#pragma unroll
        for (int kt = 0; kt < 4; ++kt) {
            v16bf b = ldB(wsw + OFF_WO, kt * 8 + nt);
#pragma unroll
            for (int mt = 0; mt < 6; ++mt)
                acc[mt] = wmma_acc(ldsA(sO, CC, mt * 16, kt * 32), b, acc[mt]);
        }
        int n = lane & 15, h = lane >> 4;
#pragma unroll
        for (int mt = 0; mt < 6; ++mt)
#pragma unroll
            for (int r = 0; r < 8; ++r) {
                int row = mt * 16 + r + 8 * h;       // = g*32 + t
                int g = row / TOKS;
                int c = nt * 16 + n;
                float att = acc[mt][r] + bo[c];
                sXg[row * CC + c] += gamma[g * CC + c] * att;
            }
    }
    __syncthreads();

    // ---- Stage F: ada_ln2 -> xn2 bf16 [t][g*128+c] (reuses union region).
    {
        int t = tid >> 3, i = tid & 7;
        float s = 0.f, s2 = 0.f;
#pragma unroll
        for (int g = 0; g < GG; ++g)
#pragma unroll
            for (int j = 0; j < 16; ++j) {
                float v = sXg[(g * TOKS + t) * CC + i * 16 + j];
                s += v; s2 += v * v;
            }
#pragma unroll
        for (int off = 4; off >= 1; off >>= 1) {
            s  += __shfl_xor(s,  off, 8);
            s2 += __shfl_xor(s2, off, 8);
        }
        float mu   = s * (1.f / 384.f);
        float var  = s2 * (1.f / 384.f) - mu * mu;
        float rstd = rsqrtf(var + 1e-5f);
#pragma unroll
        for (int g = 0; g < GG; ++g)
#pragma unroll
            for (int j = 0; j < 16; ++j) {
                int c = i * 16 + j;
                float v  = sXg[(g * TOKS + t) * CC + c];
                float sc = mod[768 + g * 256 + c];
                float sh = mod[768 + g * 256 + 128 + c];
                sXn2[t * (GG * CC) + g * CC + c] =
                    (__bf16)(((v - mu) * rstd) * (1.f + sc) + sh);
            }
    }
    __syncthreads();

    // ---- Stage G: MLP1, h = silu(xn2(32x384) @ W1 + b1) -> bf16 in sH.
    {
        int nt = wave;
        v8f a0 = {}, a1 = {};
        for (int kt = 0; kt < 12; ++kt) {            // unroll: compiler's choice
            v16bf b = ldB(wsw + OFF_W1, kt * 8 + nt);
            a0 = wmma_acc(ldsA(sXn2, GG * CC, 0,  kt * 32), b, a0);
            a1 = wmma_acc(ldsA(sXn2, GG * CC, 16, kt * 32), b, a1);
        }
        int n = lane & 15, h = lane >> 4;
#pragma unroll
        for (int mt = 0; mt < 2; ++mt) {
            v8f acc = mt ? a1 : a0;
#pragma unroll
            for (int r = 0; r < 8; ++r) {
                int t = mt * 16 + r + 8 * h;
                int c = nt * 16 + n;
                float x  = acc[r] + b1[c];
                float sl = x / (1.f + __expf(-x));
                sH[t * CC + c] = (__bf16)sl;
            }
        }
    }
    __syncthreads();

    // ---- Stage H: MLP2 + final residual -> global out.
    {
        int nt = wave;
        v8f a0 = {}, a1 = {};
#pragma unroll
        for (int kt = 0; kt < 4; ++kt) {
            v16bf b = ldB(wsw + OFF_W2, kt * 8 + nt);
            a0 = wmma_acc(ldsA(sH, CC, 0,  kt * 32), b, a0);
            a1 = wmma_acc(ldsA(sH, CC, 16, kt * 32), b, a1);
        }
        int n = lane & 15, h = lane >> 4;
#pragma unroll
        for (int mt = 0; mt < 2; ++mt) {
            v8f acc = mt ? a1 : a0;
#pragma unroll
            for (int r = 0; r < 8; ++r) {
                int t = mt * 16 + r + 8 * h;
                int c = nt * 16 + n;
                float y = acc[r] + b2[c];
                out[(size_t)(tok0 + t) * CC + c] =
                    sSkip[t * CC + c] + gamma_mlp[c] * y;
            }
        }
    }

    // ---- mask_out = mask0 & mask1 & mask2 (as float), appended after x.
    if (tid < TOKS) {
        int tok = tok0 + tid;
        bool m = mk0[tok] && mk1[tok] && mk2[tok];
        out[(size_t)T * CC + tok] = m ? 1.f : 0.f;
    }
}

// ---------------------------------------------------------------------------
extern "C" void kernel_launch(void* const* d_in, const int* in_sizes, int n_in,
                              void* d_out, int out_size, void* d_ws, size_t ws_size,
                              hipStream_t stream) {
    (void)in_sizes; (void)n_in; (void)out_size; (void)ws_size;
    const float* x0        = (const float*)d_in[0];
    const float* x1        = (const float*)d_in[1];
    const float* x2        = (const float*)d_in[2];
    const unsigned char* m0 = (const unsigned char*)d_in[3];
    const unsigned char* m1 = (const unsigned char*)d_in[4];
    const unsigned char* m2 = (const unsigned char*)d_in[5];
    const int*   gidx      = (const int*)d_in[6];
    const float* W_proj0   = (const float*)d_in[7];
    const float* W_proj2   = (const float*)d_in[8];
    const float* emb       = (const float*)d_in[9];
    const float* ln1_W     = (const float*)d_in[10];
    const float* ln1_b     = (const float*)d_in[11];
    const float* ln2_W     = (const float*)d_in[12];
    const float* ln2_b     = (const float*)d_in[13];
    const float* Wq        = (const float*)d_in[14];
    const float* Wk        = (const float*)d_in[15];
    const float* Wv        = (const float*)d_in[16];
    const float* Wo        = (const float*)d_in[17];
    const float* bo        = (const float*)d_in[18];
    const float* W_skip    = (const float*)d_in[19];
    const float* b_skip    = (const float*)d_in[20];
    const float* gamma     = (const float*)d_in[21];
    const float* gamma_mlp = (const float*)d_in[22];
    const float* W1        = (const float*)d_in[23];
    const float* b1        = (const float*)d_in[24];
    const float* W2        = (const float*)d_in[25];
    const float* b2        = (const float*)d_in[26];

    __bf16* wsw = (__bf16*)d_ws;
    float*  mod = (float*)((char*)d_ws + WS_MOD_BYTE_OFF);

    const int T = 2 * 4096 * 8;

    k_mod<<<6, 256, 0, stream>>>(emb, gidx, ln1_W, ln1_b, ln2_W, ln2_b, mod);
    k_swz<<<448, 512, 0, stream>>>(W_proj0, W_proj2, W_skip, Wq, Wk, Wv, Wo,
                                   W1, W2, wsw);
    k_fused<<<T / TOKS, 256, 0, stream>>>(x0, x1, x2, m0, m1, m2, wsw, mod,
                                          bo, b_skip, gamma, gamma_mlp, b1, b2,
                                          (float*)d_out, T);
}
